// Attention_39883066311345
// MI455X (gfx1250) — matrix operand order repaired and checked by reading, not compiled, not run on hardware
//
#include <hip/hip_runtime.h>
#include <hip/hip_fp16.h>

typedef __attribute__((ext_vector_type(16))) _Float16 v16h;
typedef __attribute__((ext_vector_type(8)))  float    v8f;

// Problem constants (fixed by the reference harness).
constexpr int B_  = 64;
constexpr int N_  = 256;
constexpr int C_  = 512;          // INP == OUP
constexpr int H_  = 16;
constexpr int D_  = 32;           // head dim
constexpr int QKVN = 3 * C_;      // 1536
constexpr float SCALE = 0.17677669529663687f;  // 32^-0.5

// ---------------------------------------------------------------------------
// WMMA helpers (wave32, V_WMMA_F32_16X16X32_F16).
// A fragment (16x32 f16, M x K): lane (h=lane>>4, m=lane&15) holds row m,
//   a[i]   = A[m][k0 + h*8 + i]      (i = 0..7)
//   a[8+i] = A[m][k0 + 16 + h*8 + i]
// B fragment (32x16 f16, K x N): lane holds column n = lane&15,
//   b[i]   = B[k0 + h*8 + i][n]      (i = 0..7)
//   b[8+i] = B[k0 + 16 + h*8 + i][n]
// C/D (16x16 f32): c[r] = C[(lane>>4)*8 + r][lane&15].
// ---------------------------------------------------------------------------
__device__ inline v8f wmma_f16(v16h a, v16h b, v8f c) {
  return __builtin_amdgcn_wmma_f32_16x16x32_f16(
      /*neg_a=*/false, a, /*neg_b=*/false, b,
      /*c_mod=*/(short)0, c, /*reuse_a=*/false, /*reuse_b=*/false);
}

// Raw fragment loads from a per-lane base pointer (all lane math pre-hoisted).
__device__ inline v16h ld_a(const _Float16* p) {  // two 16B chunks, 32B apart
  v16h a;
#pragma unroll
  for (int i = 0; i < 8; ++i) { a[i] = p[i]; a[8 + i] = p[16 + i]; }
  return a;
}
__device__ inline v16h ld_b(const _Float16* p) {  // two 16B chunks, 32B apart
  v16h b;
#pragma unroll
  for (int i = 0; i < 8; ++i) { b[i] = p[i]; b[8 + i] = p[16 + i]; }
  return b;
}

// Per-lane base offsets (element units) for fragment streams.
__device__ inline size_t a_base(int row, int ld, int lane) {
  return (size_t)(row + (lane & 15)) * ld + ((lane >> 4) & 1) * 8;
}
__device__ inline size_t b_base(int col, int ld, int lane) {
  return (size_t)(col + (lane & 15)) * ld + ((lane >> 4) & 1) * 8;
}

// ---------------------------------------------------------------------------
// Kernel 1: fp32 -> f16 conversions (+ transpose of w_qkv to output-major).
// ---------------------------------------------------------------------------
__global__ void k_convert(const float* __restrict__ x,
                          const float* __restrict__ wqkv,
                          const float* __restrict__ wout,
                          _Float16* __restrict__ Xh,
                          _Float16* __restrict__ Wqkvt,
                          _Float16* __restrict__ Wouth) {
  const long stride = (long)gridDim.x * blockDim.x;
  const long tid0 = (long)blockIdx.x * blockDim.x + threadIdx.x;

  const long nx = (long)B_ * N_ * C_;
  for (long t = tid0; t < nx; t += stride) Xh[t] = (_Float16)x[t];

  const long nq = (long)C_ * QKVN;
  for (long t = tid0; t < nq; t += stride) {
    const long r = t / QKVN, c = t % QKVN;       // wqkv[r][c]
    Wqkvt[c * C_ + r] = (_Float16)wqkv[t];       // -> [1536][512]
  }

  const long nw = (long)C_ * C_;
  for (long t = tid0; t < nw; t += stride) Wouth[t] = (_Float16)wout[t];  // [p][o]
}

// ---------------------------------------------------------------------------
// Kernel 2: QKV projection GEMM, register-blocked 32(M) x 64(N) per wave:
// 2 A-fragments x 4 B-fragments -> 8 WMMAs per K-step from 6 x 32B loads.
// All fragment streams addressed off TWO per-lane base pointers with
// compile-time immediate offsets (16*C_, ...) to keep address VGPRs minimal.
// ---------------------------------------------------------------------------
__global__ void __launch_bounds__(256, 1)
k_qkv(const _Float16* __restrict__ Xh,
      const _Float16* __restrict__ Wqkvt,
      _Float16* __restrict__ Q,
      _Float16* __restrict__ K,
      _Float16* __restrict__ V) {
  const int lane = threadIdx.x & 31;
  const int wave = threadIdx.x >> 5;
  const int wg   = blockIdx.x * (blockDim.x >> 5) + wave;
  const int tilesN = QKVN / 64;                  // 24
  const int tm = (wg / tilesN) * 32;             // < 16384
  const int tn = (wg % tilesN) * 64;             // < 1536

  const _Float16* pA = Xh    + a_base(tm, C_, lane);
  const _Float16* pB = Wqkvt + b_base(tn, C_, lane);

  v8f acc[2][4] = {};
#pragma unroll 2
  for (int k0 = 0; k0 < C_; k0 += 32) {
    __builtin_prefetch(pA + k0 + 128, 0, 1);
    v16h a0 = ld_a(pA + k0);
    v16h a1 = ld_a(pA + 16 * C_ + k0);
#pragma unroll
    for (int nf = 0; nf < 4; ++nf) {
      v16h b = ld_b(pB + nf * 16 * C_ + k0);
      acc[0][nf] = wmma_f16(a0, b, acc[0][nf]);
      acc[1][nf] = wmma_f16(a1, b, acc[1][nf]);
    }
  }

  const int halfg = (lane >> 4) & 1;
#pragma unroll
  for (int nf = 0; nf < 4; ++nf) {
    const int j   = tn + nf * 16 + (lane & 15);
    const int sel = j >> 9;            // 0=q, 1=k, 2=v
    const int c   = j & 511;
    const int hh  = c >> 5, d = c & 31;
    _Float16* dst = (sel == 0) ? Q : (sel == 1 ? K : V);
    const float mul = (sel == 0) ? SCALE : 1.0f;
#pragma unroll
    for (int mf = 0; mf < 2; ++mf) {
#pragma unroll
      for (int r = 0; r < 8; ++r) {
        const int row = tm + mf * 16 + halfg * 8 + r;  // token index in [0, B*N)
        const int bb = row >> 8, n = row & 255;
        dst[(((size_t)(bb * H_ + hh) * N_) + n) * D_ + d] =
            (_Float16)(acc[mf][nf][r] * mul);
      }
    }
  }
}

// ---------------------------------------------------------------------------
// Kernel 3: attention for one (b, h) per block. 128 threads = 4 waves,
// each wave owns a 16-row strip; 4 strip iterations cover N = 256.
// LDS budget: 16KB (K) + 16KB (V^T) + 32KB (P) = 64KB exactly.
// ---------------------------------------------------------------------------
__global__ void __launch_bounds__(128, 1)
k_attn(const _Float16* __restrict__ Qg,
       const _Float16* __restrict__ Kg,
       const _Float16* __restrict__ Vg,
       const float* __restrict__ table,
       _Float16* __restrict__ AO) {
  __shared__ _Float16 Ksh[N_ * D_];     // [256][32] row-major
  __shared__ _Float16 Vt[D_ * N_];      // [32][256] (transposed -> B-frag contiguous)
  __shared__ _Float16 Psh[4 * 16 * N_]; // per-wave [16][256]

  const int bh = blockIdx.x;
  const int b = bh >> 4, h = bh & 15;
  const _Float16* Qb = Qg + (size_t)bh * N_ * D_;
  const _Float16* Kb = Kg + (size_t)bh * N_ * D_;
  const _Float16* Vb = Vg + (size_t)bh * N_ * D_;

  const int tid = threadIdx.x;
  const int lane = tid & 31, wave = tid >> 5;
  const int halfg = (lane >> 4) & 1, nlo = lane & 15;

  // Stage K (row-major) and V^T into LDS.
  for (int r = tid; r < N_; r += 128) {
#pragma unroll
    for (int d = 0; d < D_; ++d) {
      _Float16 kv = Kb[r * D_ + d];
      Ksh[r * D_ + d] = kv;
      Vt[d * N_ + r]  = Vb[r * D_ + d];
    }
  }
  __syncthreads();

  _Float16* Pw = Psh + wave * 16 * N_;

  for (int rb = 0; rb < 4; ++rb) {
    const int rowbase = rb * 64 + wave * 16;

    // ---- S = (scaled Q) K^T : one WMMA per 16-col tile (K-dim == D == 32)
    v16h aq = ld_a(Qb + a_base(rowbase, D_, lane));
    v8f s[16];
#pragma unroll
    for (int ct = 0; ct < 16; ++ct) {
      // column m of K^T = row m of K (row-major in LDS -> contiguous)
      v16h bk = ld_b(Ksh + b_base(ct * 16, D_, lane));
      v8f z = {};
      s[ct] = wmma_f16(aq, bk, z);
    }

    // ---- add relative position bias (inline gather; table is L2-resident)
#pragma unroll
    for (int ct = 0; ct < 16; ++ct) {
      const int j = ct * 16 + nlo;
      const int yj = j >> 4, xj = j & 15;
#pragma unroll
      for (int r = 0; r < 8; ++r) {
        const int i = rowbase + halfg * 8 + r;
        const int yi = i >> 4, xi = i & 15;
        const int idx = (yi - yj + 15) * 31 + (xi - xj + 15);
        s[ct][r] += table[idx * H_ + h];
      }
    }

    // ---- softmax over the 256 columns (rows live in regs + 16 lanes)
    float rinv[8];
#pragma unroll
    for (int r = 0; r < 8; ++r) {
      float m = s[0][r];
#pragma unroll
      for (int ct = 1; ct < 16; ++ct) m = fmaxf(m, s[ct][r]);
#pragma unroll
      for (int off = 1; off < 16; off <<= 1) m = fmaxf(m, __shfl_xor(m, off, 32));
      float sum = 0.f;
#pragma unroll
      for (int ct = 0; ct < 16; ++ct) {
        const float e = __expf(s[ct][r] - m);
        s[ct][r] = e;
        sum += e;
      }
#pragma unroll
      for (int off = 1; off < 16; off <<= 1) sum += __shfl_xor(sum, off, 32);
      rinv[r] = 1.0f / sum;
    }

    // ---- stash unnormalized P (f16) to LDS to re-fragment as an A matrix
#pragma unroll
    for (int ct = 0; ct < 16; ++ct)
#pragma unroll
      for (int r = 0; r < 8; ++r)
        Pw[(halfg * 8 + r) * N_ + ct * 16 + nlo] = (_Float16)s[ct][r];
    __syncthreads();

    // ---- O = P @ V, normalize rows, store to AO[B*N][512] f16
    const _Float16* pAP = Pw + a_base(0, N_, lane);
#pragma unroll
    for (int dt = 0; dt < 2; ++dt) {
      const _Float16* pBV = Vt + b_base(dt * 16, N_, lane);
      v8f o = {};
#pragma unroll
      for (int k0 = 0; k0 < N_; k0 += 32)
        o = wmma_f16(ld_a(pAP + k0), ld_b(pBV + k0), o);
#pragma unroll
      for (int r = 0; r < 8; ++r) {
        const int n = rowbase + halfg * 8 + r;
        const int col = dt * 16 + nlo;
        AO[((size_t)(b * N_ + n)) * C_ + h * D_ + col] = (_Float16)(o[r] * rinv[r]);
      }
    }
    __syncthreads();
  }
}

// ---------------------------------------------------------------------------
// Kernel 4: output projection out = AO @ w_out^T + b_out (fp32 result),
// register-blocked 32(M) x 64(N) per wave like k_qkv.
// B[k=o][n=p] = w_out[p][o] -> Wouth in original [p][o] layout is already
// column-contiguous for the B fragment.
// ---------------------------------------------------------------------------
__global__ void __launch_bounds__(256, 1)
k_proj(const _Float16* __restrict__ AO,
       const _Float16* __restrict__ Wouth,
       const float* __restrict__ bout,
       float* __restrict__ out) {
  const int lane = threadIdx.x & 31;
  const int wave = threadIdx.x >> 5;
  const int wg   = blockIdx.x * (blockDim.x >> 5) + wave;
  const int tilesN = C_ / 64;                    // 8
  const int tm = (wg / tilesN) * 32;
  const int tn = (wg % tilesN) * 64;

  const _Float16* pA = AO    + a_base(tm, C_, lane);
  const _Float16* pB = Wouth + b_base(tn, C_, lane);

  v8f acc[2][4] = {};
#pragma unroll 2
  for (int k0 = 0; k0 < C_; k0 += 32) {
    __builtin_prefetch(pA + k0 + 128, 0, 1);
    v16h a0 = ld_a(pA + k0);
    v16h a1 = ld_a(pA + 16 * C_ + k0);
#pragma unroll
    for (int nf = 0; nf < 4; ++nf) {
      v16h b = ld_b(pB + nf * 16 * C_ + k0);
      acc[0][nf] = wmma_f16(a0, b, acc[0][nf]);
      acc[1][nf] = wmma_f16(a1, b, acc[1][nf]);
    }
  }

  const int halfg = (lane >> 4) & 1;
#pragma unroll
  for (int nf = 0; nf < 4; ++nf) {
    const int p = tn + nf * 16 + (lane & 15);
    const float bo = bout[p];
#pragma unroll
    for (int mf = 0; mf < 2; ++mf) {
#pragma unroll
      for (int r = 0; r < 8; ++r) {
        const int row = tm + mf * 16 + halfg * 8 + r;
        out[(size_t)row * C_ + p] = acc[mf][nf][r] + bo;
      }
    }
  }
}

// ---------------------------------------------------------------------------
// Host-side launcher.
// ---------------------------------------------------------------------------
extern "C" void kernel_launch(void* const* d_in, const int* in_sizes, int n_in,
                              void* d_out, int out_size, void* d_ws, size_t ws_size,
                              hipStream_t stream) {
  const float* x     = (const float*)d_in[0];
  const float* wqkv  = (const float*)d_in[1];
  const float* table = (const float*)d_in[2];
  const float* wout  = (const float*)d_in[3];
  const float* bout  = (const float*)d_in[4];
  float* out = (float*)d_out;

  char* ws = (char*)d_ws;
  size_t off = 0;
  auto alloc = [&](size_t bytes) -> void* {
    void* p = ws + off;
    off = (off + bytes + 255) & ~(size_t)255;
    return p;
  };

  const size_t tokBytes = (size_t)B_ * N_ * C_ * sizeof(_Float16);  // 16.8 MB
  _Float16* Xh    = (_Float16*)alloc(tokBytes);        // also reused as AO
  _Float16* Wqkvt = (_Float16*)alloc((size_t)QKVN * C_ * sizeof(_Float16));
  _Float16* Wouth = (_Float16*)alloc((size_t)C_ * C_ * sizeof(_Float16));
  _Float16* Q     = (_Float16*)alloc(tokBytes);
  _Float16* K     = (_Float16*)alloc(tokBytes);
  _Float16* V     = (_Float16*)alloc(tokBytes);
  _Float16* AO    = Xh;  // Xh is dead after k_qkv; alias to save workspace.

  // 1) convert / transpose
  k_convert<<<1024, 256, 0, stream>>>(x, wqkv, wout, Xh, Wqkvt, Wouth);

  // 2) QKV GEMM: (16384/32) x (1536/64) = 512 x 24 wave-tiles, 8 waves/block
  {
    const int waves = (B_ * N_ / 32) * (QKVN / 64);   // 12288
    k_qkv<<<waves / 8, 256, 0, stream>>>(Xh, Wqkvt, Q, K, V);
  }

  // 3) attention: one block per (b, h)
  k_attn<<<B_ * H_, 128, 0, stream>>>(Q, K, V, table, AO);

  // 4) output projection: (16384/32) x (512/64) = 512 x 8 wave-tiles
  {
    const int waves = (B_ * N_ / 32) * (C_ / 64);     // 4096
    k_proj<<<waves / 8, 256, 0, stream>>>(AO, Wouth, bout, out);
  }
}
